// SelfAttention_22419729285196
// MI455X (gfx1250) — matrix operand order repaired and checked by reading, not compiled, not run on hardware
//
#include <hip/hip_runtime.h>
#include <hip/hip_bf16.h>

// ---------------- problem constants (fixed by the reference) ----------------
#define NB      4
#define SEQ     2048
#define DMODEL  1024
#define NHEADS  16
#define HD      64
#define MROWS   (NB * SEQ)          // 8192 token rows

typedef __attribute__((ext_vector_type(16))) __bf16 v16bf;
typedef __attribute__((ext_vector_type(8)))  __bf16 v8bf;
typedef __attribute__((ext_vector_type(8)))  float  v8f;

#define WMMA_BF16(A, B, C) \
    __builtin_amdgcn_wmma_f32_16x16x32_bf16(false, (A), false, (B), (short)0, (C), false, false)

// ---------------- WMMA operand loaders (CDNA5 16-bit layouts, §7.12.2) ------
// A 16x32 (MxK): lane row m = lane&15; elem e: k = (e>>3)*16 + (lane>>4)*8 + (e&7)
// -> two contiguous 8-elem (16B) chunks at kbase + half*8 and kbase + 16 + half*8.
__device__ __forceinline__ v16bf ld_a(const __bf16* row, int kbase, int lhalf) {
    v8bf c0 = *(const v8bf*)(row + kbase + lhalf * 8);
    v8bf c1 = *(const v8bf*)(row + kbase + 16 + lhalf * 8);
    v16bf r;
#pragma unroll
    for (int i = 0; i < 8; ++i) { r[i] = c0[i]; r[i + 8] = c1[i]; }
    return r;
}

__device__ __forceinline__ v16bf ld_a_scaled(const __bf16* row, int kbase, int lhalf, float s) {
    v8bf c0 = *(const v8bf*)(row + kbase + lhalf * 8);
    v8bf c1 = *(const v8bf*)(row + kbase + 16 + lhalf * 8);
    v16bf r;
#pragma unroll
    for (int i = 0; i < 8; ++i) {
        r[i]     = (__bf16)((float)c0[i] * s);
        r[i + 8] = (__bf16)((float)c1[i] * s);
    }
    return r;
}

// B 32x16 (KxN): lane col n = lane&15; elem e: k = (e>>3)*16 + (lane>>4)*8 + (e&7)
// -> two contiguous 8-elem (16B) chunks at kbase + half*8 and kbase + 16 + half*8.
__device__ __forceinline__ v16bf ld_b(const __bf16* row, int kbase, int lhalf) {
    const __bf16* p = row + kbase + lhalf * 8;
    v8bf c0 = *(const v8bf*)(p);
    v8bf c1 = *(const v8bf*)(p + 16);
    v16bf r;
#pragma unroll
    for (int i = 0; i < 8; ++i) { r[i] = c0[i]; r[i + 8] = c1[i]; }
    return r;
}
// C/D 16x16 f32: lane col n = lane&15; VGPR r holds row m = r + 8*(lane>>4).

// ---------------- kernel 1: f32 -> bf16 cast --------------------------------
__global__ void cast_f32_bf16(const float* __restrict__ src, __bf16* __restrict__ dst, int n) {
    for (int i = blockIdx.x * blockDim.x + threadIdx.x; i < n; i += gridDim.x * blockDim.x)
        dst[i] = (__bf16)src[i];
}

// ---------------- kernel 2: NT GEMM  C[MxN] = A[MxK] * B[NxK]^T -------------
// mode 0: store bf16 row-major [M,N]            (Q, K projections)
// mode 1: store bf16 transposed-per-head [n,h,d,s]  (V projection)
// mode 2: store f32 row-major + bias            (output projection)
__global__ __launch_bounds__(256) void gemm_nt_kernel(
    const __bf16* __restrict__ A, const __bf16* __restrict__ B,
    __bf16* __restrict__ Cb, float* __restrict__ Cf,
    const float* __restrict__ bias, int M, int N, int K, int mode)
{
    const int lane  = threadIdx.x & 31;
    const int l16   = lane & 15;
    const int lhalf = lane >> 4;
    const int wave  = threadIdx.x >> 5;
    const int wm    = wave >> 1;     // 0..3
    const int wn    = wave & 1;      // 0..1
    const int m0    = blockIdx.y * 128 + wm * 32;
    const int n0    = blockIdx.x * 64 + wn * 32;

    const __bf16* a0p = A + (size_t)(m0 + l16) * K;
    const __bf16* a1p = A + (size_t)(m0 + 16 + l16) * K;
    const __bf16* b0p = B + (size_t)(n0 + l16) * K;
    const __bf16* b1p = B + (size_t)(n0 + 16 + l16) * K;

    v8f c00 = {}, c01 = {}, c10 = {}, c11 = {};

    for (int k = 0; k < K; k += 32) {
        __builtin_prefetch(a0p + k + 128, 0, 0);   // global_prefetch_b8
        __builtin_prefetch(b0p + k + 128, 0, 0);
        v16bf a0 = ld_a(a0p, k, lhalf);
        v16bf a1 = ld_a(a1p, k, lhalf);
        v16bf b0 = ld_b(b0p, k, lhalf);
        v16bf b1 = ld_b(b1p, k, lhalf);
        c00 = WMMA_BF16(a0, b0, c00);
        c01 = WMMA_BF16(a0, b1, c01);
        c10 = WMMA_BF16(a1, b0, c10);
        c11 = WMMA_BF16(a1, b1, c11);
    }

    v8f cc[2][2] = {{c00, c01}, {c10, c11}};
#pragma unroll
    for (int ti = 0; ti < 2; ++ti)
#pragma unroll
    for (int tj = 0; tj < 2; ++tj) {
        const int mb = m0 + ti * 16;
        const int nb = n0 + tj * 16;
        const int n  = nb + l16;
        const int s0 = mb + 8 * lhalf;           // 8 consecutive rows start here
        if (mode == 0) {
#pragma unroll
            for (int r = 0; r < 8; ++r)
                Cb[(size_t)(s0 + r) * N + n] = (__bf16)cc[ti][tj][r];
        } else if (mode == 1) {
            // row i -> (batch, s); col j -> (head, d); Vt[((batch*16+h)*64+d)*2048 + s]
            v8bf pk;
#pragma unroll
            for (int r = 0; r < 8; ++r) pk[r] = (__bf16)cc[ti][tj][r];
            const int batch = s0 >> 11, s = s0 & 2047;
            const int hh = n >> 6, dd = n & 63;
            *(v8bf*)(Cb + ((size_t)((batch * NHEADS + hh) * HD + dd) << 11) + s) = pk;
        } else {
            const float bv = bias[n];
#pragma unroll
            for (int r = 0; r < 8; ++r)
                Cf[(size_t)(s0 + r) * N + n] = cc[ti][tj][r] + bv;
        }
    }
}

// ---------------- kernel 3: exclusive prefix-sum of V (the post-softmax mask)
// Vt layout [n,h,d,s]: each thread scans one contiguous length-S row.
__global__ void cumv_kernel(const __bf16* __restrict__ Vt, float* __restrict__ Cum) {
    const int row = blockIdx.x * blockDim.x + threadIdx.x;
    if (row >= NB * NHEADS * HD) return;
    const __bf16* v = Vt + (size_t)row * SEQ;
    float* c = Cum + (size_t)row * SEQ;
    float acc = 0.f;
    for (int s = 0; s < SEQ; ++s) { c[s] = acc; acc += (float)v[s]; }
}

// ---------------- kernel 4: flash-style attention + mask term ---------------
// One wave per 16 query rows of one (batch, head). Online softmax, WMMA for
// QK^T and PV; P transposed C-layout -> A-layout through LDS.
__global__ __launch_bounds__(32) void attn_kernel(
    const __bf16* __restrict__ Q, const __bf16* __restrict__ Kmat,
    const __bf16* __restrict__ Vt, const float* __restrict__ Cum,
    __bf16* __restrict__ Ao)
{
    const int lane  = threadIdx.x & 31;
    const int l16   = lane & 15;
    const int lhalf = lane >> 4;
    const int bid   = blockIdx.x;
    const int qb    = bid & 127;           // query block (16 rows)
    const int h     = (bid >> 7) & 15;
    const int n     = bid >> 11;

    const __bf16* qh = Q    + ((size_t)(n * SEQ + qb * 16)) * DMODEL + h * HD;
    const __bf16* kh = Kmat + ((size_t)n * SEQ) * DMODEL + h * HD;
    const __bf16* vh = Vt   + ((size_t)(n * NHEADS + h)) * HD * SEQ;
    const float*  ch = Cum  + ((size_t)(n * NHEADS + h)) * HD * SEQ;
    __bf16*       oh = Ao   + ((size_t)(n * SEQ + qb * 16)) * DMODEL + h * HD;

    // Q operand, attn scale 1/sqrt(64)=0.125 folded in (exact in bf16)
    const __bf16* qrow = qh + (size_t)l16 * DMODEL;
    const v16bf aq0 = ld_a_scaled(qrow, 0,  lhalf, 0.125f);
    const v16bf aq1 = ld_a_scaled(qrow, 32, lhalf, 0.125f);

    v8f oacc[4] = {};
    float mrow[8], lrow[8];
#pragma unroll
    for (int r = 0; r < 8; ++r) { mrow[r] = -1e30f; lrow[r] = 0.f; }

    __shared__ __align__(16) __bf16 plds[16][40];   // 80B row pitch, 16B aligned chunks

    for (int t0 = 0; t0 < SEQ; t0 += 32) {
        // ---- logits: two 16x16 tiles over 32 keys, K-dim = 64 head dims ----
        const __bf16* k0row = kh + (size_t)(t0 + l16) * DMODEL;
        const __bf16* k1row = kh + (size_t)(t0 + 16 + l16) * DMODEL;
        v8f s0v = {}, s1v = {};
        s0v = WMMA_BF16(aq0, ld_b(k0row, 0,  lhalf), s0v);
        s0v = WMMA_BF16(aq1, ld_b(k0row, 32, lhalf), s0v);
        s1v = WMMA_BF16(aq0, ld_b(k1row, 0,  lhalf), s1v);
        s1v = WMMA_BF16(aq1, ld_b(k1row, 32, lhalf), s1v);

        // ---- online softmax (row r lives in one VGPR across a 16-lane half)
#pragma unroll
        for (int r = 0; r < 8; ++r) {
            float tmax = fmaxf(s0v[r], s1v[r]);
#pragma unroll
            for (int off = 8; off >= 1; off >>= 1)
                tmax = fmaxf(tmax, __shfl_xor(tmax, off, 32));
            const float newm = fmaxf(mrow[r], tmax);
            const float corr = __expf(mrow[r] - newm);
            const float p0 = __expf(s0v[r] - newm);
            const float p1 = __expf(s1v[r] - newm);
            float psum = p0 + p1;
#pragma unroll
            for (int off = 8; off >= 1; off >>= 1)
                psum += __shfl_xor(psum, off, 32);
            lrow[r] = lrow[r] * corr + psum;
            mrow[r] = newm;
#pragma unroll
            for (int j = 0; j < 4; ++j) oacc[j][r] *= corr;
            // stage P (C layout) into LDS for transpose to A layout
            plds[r + 8 * lhalf][l16]      = (__bf16)p0;
            plds[r + 8 * lhalf][16 + l16] = (__bf16)p1;
        }
        __syncthreads();
        asm volatile("s_wait_dscnt 0" ::: "memory");

        // ---- read P back as a 16x32 A operand ----
        v16bf ap;
        {
            v8bf c0 = *(const v8bf*)&plds[l16][lhalf * 8];
            v8bf c1 = *(const v8bf*)&plds[l16][16 + lhalf * 8];
#pragma unroll
            for (int i = 0; i < 8; ++i) { ap[i] = c0[i]; ap[i + 8] = c1[i]; }
        }

        // ---- PV: Vt is [d][s] per head, so B operand is t-contiguous ----
#pragma unroll
        for (int j = 0; j < 4; ++j)
            oacc[j] = WMMA_BF16(ap, ld_b(vh + (size_t)(j * 16 + l16) * SEQ, t0, lhalf), oacc[j]);
        __syncthreads();
    }

    // ---- epilogue: normalize + faithful post-softmax mask term -------------
#pragma unroll
    for (int j = 0; j < 4; ++j) {
        const int d = j * 16 + l16;
#pragma unroll
        for (int r = 0; r < 8; ++r) {
            const int m  = r + 8 * lhalf;
            const int sg = qb * 16 + m;
            const float val = oacc[j][r] / lrow[r] - 1e9f * ch[(size_t)d * SEQ + sg];
            oh[(size_t)m * DMODEL + d] = (__bf16)val;
        }
    }
}

// ---------------- host-side launch ------------------------------------------
extern "C" void kernel_launch(void* const* d_in, const int* in_sizes, int n_in,
                              void* d_out, int out_size, void* d_ws, size_t ws_size,
                              hipStream_t stream) {
    (void)in_sizes; (void)n_in; (void)out_size; (void)ws_size;
    const float* x  = (const float*)d_in[0];
    const float* wq = (const float*)d_in[1];
    const float* wk = (const float*)d_in[2];
    const float* wv = (const float*)d_in[3];
    const float* wo = (const float*)d_in[4];
    const float* bo = (const float*)d_in[5];
    float* out = (float*)d_out;

    char* ws = (char*)d_ws;
    const size_t SZ_X = (size_t)MROWS * DMODEL * 2;       // 16 MiB bf16
    const size_t SZ_W = (size_t)DMODEL * DMODEL * 2;      //  2 MiB bf16
    __bf16* xb   = (__bf16*)(ws);
    __bf16* wqb  = (__bf16*)(ws + SZ_X);
    __bf16* wkb  = (__bf16*)(ws + SZ_X + 1 * SZ_W);
    __bf16* wvb  = (__bf16*)(ws + SZ_X + 2 * SZ_W);
    __bf16* wob  = (__bf16*)(ws + SZ_X + 3 * SZ_W);
    __bf16* Qb   = (__bf16*)(ws + SZ_X + 4 * SZ_W);
    __bf16* Kb   = (__bf16*)(ws + SZ_X + 4 * SZ_W + 1 * SZ_X);
    __bf16* VtB  = (__bf16*)(ws + SZ_X + 4 * SZ_W + 2 * SZ_X);
    float*  Cum  = (float*) (ws + SZ_X + 4 * SZ_W + 3 * SZ_X);
    __bf16* Attn = xb;   // x no longer needed once Q/K/V are built

    // 1) casts
    cast_f32_bf16<<<2048, 256, 0, stream>>>(x,  xb,  MROWS * DMODEL);
    cast_f32_bf16<<<1024, 256, 0, stream>>>(wq, wqb, DMODEL * DMODEL);
    cast_f32_bf16<<<1024, 256, 0, stream>>>(wk, wkb, DMODEL * DMODEL);
    cast_f32_bf16<<<1024, 256, 0, stream>>>(wv, wvb, DMODEL * DMODEL);
    cast_f32_bf16<<<1024, 256, 0, stream>>>(wo, wob, DMODEL * DMODEL);

    // 2) projections: grid = (N/64, M/128), 256 threads (8 waves of 32x32)
    dim3 ggrid(DMODEL / 64, MROWS / 128);
    gemm_nt_kernel<<<ggrid, 256, 0, stream>>>(xb, wqb, Qb,  nullptr, nullptr,
                                              MROWS, DMODEL, DMODEL, 0);
    gemm_nt_kernel<<<ggrid, 256, 0, stream>>>(xb, wkb, Kb,  nullptr, nullptr,
                                              MROWS, DMODEL, DMODEL, 0);
    gemm_nt_kernel<<<ggrid, 256, 0, stream>>>(xb, wvb, VtB, nullptr, nullptr,
                                              MROWS, DMODEL, DMODEL, 1);

    // 3) exclusive prefix-sum of V rows (post-softmax additive-mask term)
    cumv_kernel<<<(NB * NHEADS * HD + 255) / 256, 256, 0, stream>>>(VtB, Cum);

    // 4) attention: one wave per 16 query rows per (batch, head)
    attn_kernel<<<NB * NHEADS * (SEQ / 16), 32, 0, stream>>>(Qb, Kb, VtB, Cum, Attn);

    // 5) output projection + bias -> f32 d_out
    gemm_nt_kernel<<<ggrid, 256, 0, stream>>>(Attn, wob, nullptr, out, bo,
                                              MROWS, DMODEL, DMODEL, 2);
}
